// MultiQueryAttention_8005819039811
// MI455X (gfx1250) — matrix operand order repaired and checked by reading, not compiled, not run on hardware
//
#include <hip/hip_runtime.h>
#include <hip/hip_bf16.h>
#include <math.h>

// MQA forward for MI455X (gfx1250, wave32).
// Pipeline: x->bf16, W->W^T bf16, QKV GEMM (bf16 WMMA, f32 accum, async-LDS
// double-buffered staging), RoPE+split, flash attention (one wave per
// 16-query tile, S^T trick), output GEMM.

typedef __attribute__((ext_vector_type(16))) __bf16 v16bf;
typedef __attribute__((ext_vector_type(8)))  __bf16 v8bf;
typedef __attribute__((ext_vector_type(4)))  __bf16 v4bf;
typedef __attribute__((ext_vector_type(8)))  float  v8f;
typedef __attribute__((ext_vector_type(4)))  float  v4f;

#define WMMA_BF16(a, b, c) \
  __builtin_amdgcn_wmma_f32_16x16x32_bf16(false, (a), false, (b), (short)0, (c), false, false)

static __device__ __forceinline__ v16bf cat8(v8bf lo, v8bf hi) {
  return __builtin_shufflevector(lo, hi, 0,1,2,3,4,5,6,7,8,9,10,11,12,13,14,15);
}

// CDNA5 async global->LDS copy (16B per lane), tracked by ASYNCcnt.
// vdst holds the per-lane LDS byte address; vaddr is the 64-bit global address.
static __device__ __forceinline__ void async_ld16(unsigned int ldsOff, const void* g) {
  asm volatile("global_load_async_to_lds_b128 %0, %1, off"
               :: "v"(ldsOff), "v"((unsigned long long)(uintptr_t)g) : "memory");
}
static __device__ __forceinline__ void wait_async0() {
  asm volatile("s_wait_asynccnt 0x0" ::: "memory");
}
// Generic LDS pointer -> LDS byte offset (ISA 10.2: LDS aperture truncates to addr[31:0]).
static __device__ __forceinline__ unsigned int lds_off(const void* p) {
  return (unsigned int)(uintptr_t)p;
}

// ---------------- fp32 -> bf16 (vectorized) ----------------
__global__ void f32_to_bf16_vec4(const float* __restrict__ src,
                                 __bf16* __restrict__ dst, int n4) {
  int i = blockIdx.x * blockDim.x + threadIdx.x;
  if (i < n4) {
    v4f v = *(const v4f*)(src + (size_t)i * 4);
    v4bf o;
    o[0] = (__bf16)v[0]; o[1] = (__bf16)v[1];
    o[2] = (__bf16)v[2]; o[3] = (__bf16)v[3];
    *(v4bf*)(dst + (size_t)i * 4) = o;
  }
}

// ---------------- W (KxN f32) -> W^T (NxK bf16) ----------------
__global__ void transpose_f32_bf16(const float* __restrict__ src,
                                   __bf16* __restrict__ dst, int K, int N) {
  __shared__ float t[32][33];
  int kb = blockIdx.x * 32, nb = blockIdx.y * 32;
  int tx = threadIdx.x, ty = threadIdx.y;
  for (int i = 0; i < 32; i += 8)
    t[ty + i][tx] = src[(size_t)(kb + ty + i) * N + nb + tx];
  __syncthreads();
  for (int i = 0; i < 32; i += 8)
    dst[(size_t)(nb + ty + i) * K + kb + tx] = (__bf16)t[tx][ty + i];
}

// ---------------- pack bq|bk|bv -> 2304 floats ----------------
__global__ void pack_bias(const float* __restrict__ bq, const float* __restrict__ bk,
                          const float* __restrict__ bv, float* __restrict__ out) {
  int i = blockIdx.x * blockDim.x + threadIdx.x;
  if (i < 2048)      out[i] = bq[i];
  else if (i < 2176) out[i] = bk[i - 2048];
  else if (i < 2304) out[i] = bv[i - 2176];
}

// ---------------- bf16 GEMM: C(MxN,f32) = A(MxK) * Bt(NxK)^T + bias ----------
// 128x128 block tile, 256 threads = 8 waves, each wave 32x64 (2x4 WMMA accs).
// Double-buffered LDS staging via CDNA5 async global->LDS loads (ASYNCcnt).
__global__ __launch_bounds__(256) void gemm_bf16(
    const __bf16* __restrict__ A, const __bf16* __restrict__ Bt,
    const float* __restrict__ bias, float* __restrict__ C,
    int M, int N, int K) {
  __shared__ __align__(32) __bf16 sA[2][128 * 32];
  __shared__ __align__(32) __bf16 sB[2][128 * 32];
  const int tid   = threadIdx.x;
  const int mBase = blockIdx.x * 128;
  const int nBase = blockIdx.y * 128;
  const int wave  = tid >> 5;
  const int lane  = tid & 31;
  const int half  = lane >> 4;     // 0: lanes 0-15, 1: lanes 16-31
  const int l     = lane & 15;
  const int wm    = (wave & 3) * 32;   // wave M offset (4 waves)
  const int wn    = (wave >> 2) * 64;  // wave N offset (2 waves)

  const v8f vzero = {0.f,0.f,0.f,0.f,0.f,0.f,0.f,0.f};
  v8f acc[2][4];
  for (int i = 0; i < 2; ++i)
    for (int j = 0; j < 4; ++j) acc[i][j] = vzero;

  // Issue one 128x32 A-tile + 128x32 B-tile as async 16B transfers (4/thread).
  auto issueTile = [&](int k0, int buf) {
    for (int i = 0; i < 2; ++i) {
      int idx = tid + i * 256;          // 512 chunks of 8 bf16 per tile
      int row = idx >> 2;
      int col = (idx & 3) << 3;
      async_ld16(lds_off(&sA[buf][idx * 8]),
                 A  + (size_t)(mBase + row) * K + k0 + col);
      async_ld16(lds_off(&sB[buf][idx * 8]),
                 Bt + (size_t)(nBase + row) * K + k0 + col);
    }
  };

  const int T = K >> 5;
  issueTile(0, 0);
  for (int j = 0; j < T; ++j) {
    wait_async0();          // this wave's tile-j transfers landed in LDS
    __syncthreads();        // all waves: tile j visible; tile j-1 reads done
    if (j + 1 < T) issueTile((j + 1) << 5, (j + 1) & 1);  // overlap with compute
    const __bf16* cA = sA[j & 1];
    const __bf16* cB = sB[j & 1];
    // B frag: lane l = column, lanes<16 K=0..15, lanes>=16 K=16..31.
    v16bf bfrag[4];
    for (int ns = 0; ns < 4; ++ns)
      bfrag[ns] = *(const v16bf*)(&cB[(wn + ns * 16 + l) * 32 + half * 16]);
    // A frag: lane l = row, lanes<16 K=0..15, lanes>=16 K=16..31.
    for (int ms = 0; ms < 2; ++ms) {
      const __bf16* ap = &cA[(wm + ms * 16 + l) * 32 + half * 16];
      v16bf afrag = cat8(*(const v8bf*)ap, *(const v8bf*)(ap + 8));
      for (int ns = 0; ns < 4; ++ns)
        acc[ms][ns] = WMMA_BF16(afrag, bfrag[ns], acc[ms][ns]);
    }
    __syncthreads();        // done reading buf[j&1] before it is overwritten
  }
  // Epilogue: D layout = VGPR r -> row r (+8 for upper half-lanes), lane = col.
  for (int ms = 0; ms < 2; ++ms)
    for (int ns = 0; ns < 4; ++ns) {
      int col = nBase + wn + ns * 16 + l;
      float bv = bias[col];
      for (int r = 0; r < 8; ++r) {
        int row = mBase + wm + ms * 16 + r + half * 8;
        C[(size_t)row * N + col] = acc[ms][ns][r] + bv;
      }
    }
}

// ---------------- RoPE + split QKV(f32 4096x2304) -> Qb, Kb, Vt(bf16) -------
// Vt stored transposed per batch: Vt[b][d][s] so PV B-fragments are contiguous.
__global__ void rope_split(const float* __restrict__ QKV, __bf16* __restrict__ Qb,
                           __bf16* __restrict__ Kb, __bf16* __restrict__ Vt) {
  const int tok = blockIdx.x;
  const int s   = tok & 2047;
  const int b   = tok >> 11;
  const int tid = threadIdx.x;
  const float* row = QKV + (size_t)tok * 2304;
  const float ln_theta_over_hd = 9.210340371976184f / 128.0f; // ln(10000)/D
  for (int p = tid; p < 1024; p += 256) {   // Q rotary pairs
    int i = p & 63;
    float inv = __expf(-(float)(2 * i) * ln_theta_over_hd);
    float ang = (float)s * inv;
    float c = __cosf(ang), sn = __sinf(ang);
    float q0 = row[2 * p], q1 = row[2 * p + 1];
    Qb[(size_t)tok * 2048 + 2 * p]     = (__bf16)(q0 * c - q1 * sn);
    Qb[(size_t)tok * 2048 + 2 * p + 1] = (__bf16)(q0 * sn + q1 * c);
  }
  if (tid < 64) {                           // K rotary pairs
    int i = tid;
    float inv = __expf(-(float)(2 * i) * ln_theta_over_hd);
    float ang = (float)s * inv;
    float c = __cosf(ang), sn = __sinf(ang);
    float k0 = row[2048 + 2 * i], k1 = row[2048 + 2 * i + 1];
    Kb[(size_t)tok * 128 + 2 * i]     = (__bf16)(k0 * c - k1 * sn);
    Kb[(size_t)tok * 128 + 2 * i + 1] = (__bf16)(k0 * sn + k1 * c);
  }
  if (tid < 128) {                          // V transpose-convert
    Vt[(size_t)b * 128 * 2048 + (size_t)tid * 2048 + s] = (__bf16)row[2176 + tid];
  }
}

// ---------------- Flash MQA attention: one wave per (b,h,16-query tile) ------
// Computes S^T = K_tile(32x128) x Q^T(128x16): the resulting D-layout accum is
// exactly the A-fragment layout for P x V (zero-shuffle softmax->WMMA handoff).
__global__ __launch_bounds__(128) void mqa_flash(
    const __bf16* __restrict__ Qb, const __bf16* __restrict__ Kb,
    const __bf16* __restrict__ Vt, __bf16* __restrict__ AO) {
  const int lane = threadIdx.x & 31;
  const int gw   = blockIdx.x * 4 + (threadIdx.x >> 5);
  const int qt   = gw & 127;
  const int h    = (gw >> 7) & 15;
  const int b    = gw >> 11;
  const int qbase = qt << 4;
  const int half  = lane >> 4;
  const int l     = lane & 15;
  const int q     = qbase + l;         // this lane's query row

  // Q^T B-fragments: lane l = column q, d contiguous per half.
  const __bf16* Qrow = Qb + (size_t)(b * 2048 + qbase + l) * 2048 + h * 128;
  v16bf qf[4];
  for (int c = 0; c < 4; ++c)
    qf[c] = *(const v16bf*)(Qrow + c * 32 + half * 16);

  const v8f vzero = {0.f,0.f,0.f,0.f,0.f,0.f,0.f,0.f};
  v8f o[8];
  for (int d = 0; d < 8; ++d) o[d] = vzero;
  float mrun = -__builtin_inff(), lsum = 0.0f;

  const __bf16* Kbase = Kb + (size_t)b * 2048 * 128;
  const __bf16* Vbase = Vt + (size_t)b * 128 * 2048;
  const int ktiles = (qbase + 47) >> 5;           // causal: k <= qbase+15
  const float scale = 0.08838834764831845f;       // 1/sqrt(128)

  for (int j = 0; j < ktiles; ++j) {
    const int kb0 = j << 5;
    if (j + 1 < ktiles) {   // prefetch next K tile (global_prefetch_b8)
      __builtin_prefetch(Kbase + (size_t)(kb0 + 32 + lane) * 128, 0, 1);
    }
    v8f sacc[2] = {vzero, vzero};
    for (int c = 0; c < 4; ++c) {
      for (int ms = 0; ms < 2; ++ms) {
        // K A-fragment: lane l = k-row, d split {0..15}/{16..31}.
        const __bf16* kp = Kbase + (size_t)(kb0 + ms * 16 + l) * 128 + c * 32 + half * 16;
        v16bf kf = cat8(*(const v8bf*)kp, *(const v8bf*)(kp + 8));
        sacc[ms] = WMMA_BF16(kf, qf[c], sacc[ms]);
      }
    }
    // Scale + causal mask + online softmax. sacc element r = k-row
    // kb0 + ms*16 + half*8 + r for query column q (= lane).
    float tmax = -__builtin_inff();
    for (int ms = 0; ms < 2; ++ms)
      for (int r = 0; r < 8; ++r) {
        float v = sacc[ms][r] * scale;
        int kk = kb0 + ms * 16 + half * 8 + r;
        if (kk > q) v = -__builtin_inff();
        sacc[ms][r] = v;
        tmax = fmaxf(tmax, v);
      }
    tmax = fmaxf(tmax, __shfl_xor(tmax, 16, 32));
    const float mnew  = fmaxf(mrun, tmax);
    const float alpha = __expf(mrun - mnew);
    mrun = mnew;
    float psum = 0.0f;
    v16bf pf;                    // P as A-fragment: direct element copy
    for (int ms = 0; ms < 2; ++ms)
      for (int r = 0; r < 8; ++r) {
        float p = __expf(sacc[ms][r] - mnew);
        psum += p;
        pf[ms * 8 + r] = (__bf16)p;
      }
    psum += __shfl_xor(psum, 16, 32);
    lsum = lsum * alpha + psum;
    // Rescale O: alpha for q-row (r + half*8) lives in lane (r + half*8).
    float arow[8];
    for (int r = 0; r < 8; ++r)
      arow[r] = __shfl(alpha, r + half * 8, 32);
    for (int dt = 0; dt < 8; ++dt)
      for (int r = 0; r < 8; ++r)
        o[dt][r] *= arow[r];
    // O += P x V; V B-fragment contiguous thanks to transposed Vt.
    for (int dt = 0; dt < 8; ++dt) {
      const __bf16* vp = Vbase + (size_t)(dt * 16 + l) * 2048 + kb0 + half * 8;
      v16bf vf = cat8(*(const v8bf*)vp, *(const v8bf*)(vp + 16));
      o[dt] = WMMA_BF16(pf, vf, o[dt]);
    }
  }
  // Normalize and store bf16 into (token, h*128+d) layout.
  float lrow[8];
  for (int r = 0; r < 8; ++r)
    lrow[r] = 1.0f / __shfl(lsum, r + half * 8, 32);
  for (int r = 0; r < 8; ++r) {
    __bf16* op = AO + (size_t)(b * 2048 + qbase + half * 8 + r) * 2048 + h * 128 + l;
    for (int dt = 0; dt < 8; ++dt)
      op[dt * 16] = (__bf16)(o[dt][r] * lrow[r]);
  }
}

// ---------------- host orchestration ----------------
extern "C" void kernel_launch(void* const* d_in, const int* in_sizes, int n_in,
                              void* d_out, int out_size, void* d_ws, size_t ws_size,
                              hipStream_t stream) {
  (void)in_sizes; (void)n_in; (void)out_size; (void)ws_size;
  const float* x  = (const float*)d_in[0];
  const float* Wq = (const float*)d_in[1];
  const float* bq = (const float*)d_in[2];
  const float* Wk = (const float*)d_in[3];
  const float* bk = (const float*)d_in[4];
  const float* Wv = (const float*)d_in[5];
  const float* bv = (const float*)d_in[6];
  const float* Wo = (const float*)d_in[7];
  const float* bo = (const float*)d_in[8];
  float* out = (float*)d_out;

  char* ws = (char*)d_ws;
  size_t off = 0;
  auto alloc = [&](size_t bytes) -> void* {
    void* p = ws + off;
    off = (off + bytes + 255) & ~(size_t)255;
    return p;
  };
  __bf16* Xb   = (__bf16*)alloc(4096ull * 2048 * 2);   // x in bf16
  __bf16* Wt   = (__bf16*)alloc(2304ull * 2048 * 2);   // [Wq|Wk|Wv]^T bf16
  __bf16* Wot  = (__bf16*)alloc(2048ull * 2048 * 2);   // Wo^T bf16
  float*  QKVf = (float*) alloc(4096ull * 2304 * 4);   // fused QKV, pre-RoPE
  __bf16* Qb   = (__bf16*)alloc(4096ull * 2048 * 2);   // roped Q
  __bf16* Kbq  = (__bf16*)alloc(4096ull * 128 * 2);    // roped K
  __bf16* Vt   = (__bf16*)alloc(2ull * 128 * 2048 * 2);// V transposed per batch
  __bf16* AO   = (__bf16*)alloc(4096ull * 2048 * 2);   // attention output
  float*  bQKV = (float*) alloc(2304ull * 4);

  f32_to_bf16_vec4<<<(4096 * 2048 / 4) / 256, 256, 0, stream>>>(x, Xb, 4096 * 2048 / 4);
  transpose_f32_bf16<<<dim3(64, 64), dim3(32, 8), 0, stream>>>(Wq, Wt, 2048, 2048);
  transpose_f32_bf16<<<dim3(64, 4),  dim3(32, 8), 0, stream>>>(Wk, Wt + 2048ull * 2048, 2048, 128);
  transpose_f32_bf16<<<dim3(64, 4),  dim3(32, 8), 0, stream>>>(Wv, Wt + 2176ull * 2048, 2048, 128);
  transpose_f32_bf16<<<dim3(64, 64), dim3(32, 8), 0, stream>>>(Wo, Wot, 2048, 2048);
  pack_bias<<<9, 256, 0, stream>>>(bq, bk, bv, bQKV);

  gemm_bf16<<<dim3(32, 18), 256, 0, stream>>>(Xb, Wt, bQKV, QKVf, 4096, 2304, 2048);
  rope_split<<<4096, 256, 0, stream>>>(QKVf, Qb, Kbq, Vt);
  mqa_flash<<<1024, 128, 0, stream>>>(Qb, Kbq, Vt, AO);
  gemm_bf16<<<dim3(32, 16), 256, 0, stream>>>(AO, Wot, bo, out, 4096, 2048, 2048);
}
